// PVConv_mink_16733192585419
// MI455X (gfx1250) — matrix operand order repaired and checked by reading, not compiled, not run on hardware
//
#include <hip/hip_runtime.h>
#include <cstdint>
#include <cstddef>

typedef __attribute__((ext_vector_type(16))) __bf16          v16bf;
typedef __attribute__((ext_vector_type(8)))  float           v8f;
typedef __attribute__((ext_vector_type(4)))  float           v4f;
typedef __attribute__((ext_vector_type(8)))  unsigned short  us8;

#define DEV static __device__ __forceinline__

namespace {
constexpr int BB = 8, CC = 32, NN = 65536, OO = 32, RR = 32, R3 = RR * RR * RR;
constexpr float EPSBN = 1e-4f, NEGS = 0.1f;
constexpr int NBLK = 256;

constexpr size_t aln(size_t x) { return (x + 255) & ~(size_t)255; }
// workspace layout
constexpr size_t OFF_NORMC = 0;                                       // [B][3][N] f32
constexpr size_t OFF_MEAN  = aln(OFF_NORMC + (size_t)BB * 3 * NN * 4);// [B][3] f32
constexpr size_t OFF_SCALE = aln(OFF_MEAN + (size_t)BB * 3 * 4);      // [B] f32
constexpr size_t OFF_PSUM  = aln(OFF_SCALE + (size_t)BB * 4);         // [B][256][3]
constexpr size_t OFF_PMAX  = aln(OFF_PSUM + (size_t)BB * NBLK * 3 * 4);
constexpr size_t OFF_INDS  = aln(OFF_PMAX + (size_t)BB * NBLK * 4);   // [B][N] i32
constexpr size_t OFF_SUMS  = aln(OFF_INDS + (size_t)BB * NN * 4);     // [B][R3][C] f32 -> avg
constexpr size_t OFF_CNT   = OFF_SUMS + (size_t)BB * R3 * CC * 4;     // [B][R3] f32 (contig w/ SUMS)
constexpr size_t OFF_AVGBF = aln(OFF_CNT + (size_t)BB * R3 * 4);      // [B][R3][C] bf16
constexpr size_t OFF_H1    = aln(OFF_AVGBF + (size_t)BB * R3 * CC * 2);
constexpr size_t OFF_H2    = aln(OFF_H1 + (size_t)BB * R3 * CC * 2);  // f32
constexpr size_t OFF_WF1   = aln(OFF_H2 + (size_t)BB * R3 * CC * 4);  // 27*2*32*16 bf16 frags
constexpr size_t OFF_WF2   = aln(OFF_WF1 + (size_t)27 * 2 * 32 * 16 * 2);
constexpr size_t OFF_PWF   = aln(OFF_WF2 + (size_t)27 * 2 * 32 * 16 * 2); // 2*2*32*16 bf16
} // namespace

DEV unsigned short f2bf(float f) {  // fp32 -> bf16, round-to-nearest-even
  unsigned u = __float_as_uint(f);
  u += 0x7fffu + ((u >> 16) & 1u);
  return (unsigned short)(u >> 16);
}

DEV void gatomic_add_f32(float* p, float v) {  // real CDNA5 fp32 atomic (no CAS loop)
  asm volatile("global_atomic_add_f32 %0, %1, off" :: "v"(p), "v"(v) : "memory");
}

DEV v8f wmma_bf16(v16bf a, v16bf b, v8f c) {
  return __builtin_amdgcn_wmma_f32_16x16x32_bf16(false, a, false, b, (short)0, c,
                                                 false, false);
}

// ---------------- coordinate statistics ----------------
__global__ void k_mean_partial(const float* __restrict__ coords, float* __restrict__ psum) {
  __shared__ float sd[3 * 256];
  int b = blockIdx.x >> 8, blk = blockIdx.x & 255;
  int n = blk * 256 + threadIdx.x;
  const float* cb = coords + (size_t)b * 3 * NN;
  sd[threadIdx.x]       = cb[n];
  sd[256 + threadIdx.x] = cb[NN + n];
  sd[512 + threadIdx.x] = cb[2 * NN + n];
  __syncthreads();
  for (int s = 128; s > 0; s >>= 1) {
    if ((int)threadIdx.x < s) {
      sd[threadIdx.x] += sd[threadIdx.x + s];
      sd[256 + threadIdx.x] += sd[256 + threadIdx.x + s];
      sd[512 + threadIdx.x] += sd[512 + threadIdx.x + s];
    }
    __syncthreads();
  }
  if (threadIdx.x == 0) {
    float* o = psum + ((size_t)b * NBLK + blk) * 3;
    o[0] = sd[0]; o[1] = sd[256]; o[2] = sd[512];
  }
}

__global__ void k_mean_final(const float* __restrict__ psum, float* __restrict__ mean) {
  __shared__ float sd[3 * 256];
  int b = blockIdx.x;
  const float* pp = psum + (size_t)b * NBLK * 3;
  sd[threadIdx.x]       = pp[threadIdx.x * 3 + 0];
  sd[256 + threadIdx.x] = pp[threadIdx.x * 3 + 1];
  sd[512 + threadIdx.x] = pp[threadIdx.x * 3 + 2];
  __syncthreads();
  for (int s = 128; s > 0; s >>= 1) {
    if ((int)threadIdx.x < s) {
      sd[threadIdx.x] += sd[threadIdx.x + s];
      sd[256 + threadIdx.x] += sd[256 + threadIdx.x + s];
      sd[512 + threadIdx.x] += sd[512 + threadIdx.x + s];
    }
    __syncthreads();
  }
  if (threadIdx.x == 0) {
    mean[b * 3 + 0] = sd[0]   * (1.0f / NN);
    mean[b * 3 + 1] = sd[256] * (1.0f / NN);
    mean[b * 3 + 2] = sd[512] * (1.0f / NN);
  }
}

__global__ void k_rad_partial(const float* __restrict__ coords, const float* __restrict__ mean,
                              float* __restrict__ pmax) {
  __shared__ float sd[256];
  int b = blockIdx.x >> 8, blk = blockIdx.x & 255;
  int n = blk * 256 + threadIdx.x;
  const float* cb = coords + (size_t)b * 3 * NN;
  float dx = cb[n] - mean[b * 3 + 0];
  float dy = cb[NN + n] - mean[b * 3 + 1];
  float dz = cb[2 * NN + n] - mean[b * 3 + 2];
  sd[threadIdx.x] = dx * dx + dy * dy + dz * dz;
  __syncthreads();
  for (int s = 128; s > 0; s >>= 1) {
    if ((int)threadIdx.x < s) sd[threadIdx.x] = fmaxf(sd[threadIdx.x], sd[threadIdx.x + s]);
    __syncthreads();
  }
  if (threadIdx.x == 0) pmax[b * NBLK + blk] = sd[0];
}

__global__ void k_rad_final(const float* __restrict__ pmax, float* __restrict__ scale) {
  __shared__ float sd[256];
  int b = blockIdx.x;
  sd[threadIdx.x] = pmax[(size_t)b * NBLK + threadIdx.x];
  __syncthreads();
  for (int s = 128; s > 0; s >>= 1) {
    if ((int)threadIdx.x < s) sd[threadIdx.x] = fmaxf(sd[threadIdx.x], sd[threadIdx.x + s]);
    __syncthreads();
  }
  if (threadIdx.x == 0) scale[b] = 2.0f * sqrtf(sd[0]);
}

// ---------------- voxelization (scatter-mean) ----------------
__global__ void k_voxelize(const float* __restrict__ coords, const float* __restrict__ features,
                           const float* __restrict__ mean, const float* __restrict__ scale,
                           float* __restrict__ normc, int* __restrict__ inds,
                           float* __restrict__ sums, float* __restrict__ cnt) {
  int gid = blockIdx.x * 256 + threadIdx.x;
  int b = gid >> 16, n = gid & 65535;
  const float* cb = coords + (size_t)b * 3 * NN;
  float inv = 1.0f / scale[b];
  int vox[3];
  int vidx = 0;
#pragma unroll
  for (int d = 0; d < 3; ++d) {
    float nc = (cb[d * NN + n] - mean[b * 3 + d]) * inv + 0.5f;
    nc = fminf(fmaxf(nc * RR, 0.0f), (float)(RR - 1));
    normc[((size_t)b * 3 + d) * NN + n] = nc;
    vox[d] = (int)rintf(nc);          // round-half-even, matches jnp.round
  }
  vidx = (vox[0] * RR + vox[1]) * RR + vox[2];
  inds[(size_t)b * NN + n] = vidx;
  float* base = sums + ((size_t)b * R3 + vidx) * CC;
#pragma unroll 4
  for (int c = 0; c < CC; ++c)
    gatomic_add_f32(base + c, features[((size_t)b * CC + c) * NN + n]);
  gatomic_add_f32(cnt + (size_t)b * R3 + vidx, 1.0f);
}

__global__ void k_finalize_avg(float* __restrict__ sums, const float* __restrict__ cnt,
                               unsigned short* __restrict__ avgbf) {
  size_t gid = (size_t)blockIdx.x * 256 + threadIdx.x;  // B*R3*C threads
  size_t v = gid >> 5;
  float c = cnt[v];
  float a = sums[gid] / fmaxf(c, 1.0f);
  sums[gid] = a;                 // avg grid (fp32, channels-last) in place
  avgbf[gid] = f2bf(a);          // bf16 copy for conv1 input
}

// ---------------- pack weights into WMMA A-fragment lane layout ----------------
// A 16x32 packing used here (contiguous k, same as B): lane<16 holds K = 0..15 (h=0..15);
// lane>=16 holds K = 16..31. Row m = mtile*16 + (lane&15).
__global__ void k_pack(const float* __restrict__ w1, const float* __restrict__ w2,
                       const float* __restrict__ pw,
                       unsigned short* __restrict__ wf1, unsigned short* __restrict__ wf2,
                       unsigned short* __restrict__ pwf) {
  int idx = blockIdx.x * 256 + threadIdx.x;
  if (idx < 3456) {  // two convs: 27 offsets * 2 mtiles * 32 lanes each
    const float* w = (idx < 1728) ? w1 : w2;
    unsigned short* wf = (idx < 1728) ? wf1 : wf2;
    int i = (idx < 1728) ? idx : idx - 1728;
    int l = i & 31, mt = (i >> 5) & 1, t = i >> 6;   // t = (kd*3+kh)*3+kw
    int m = mt * 16 + (l & 15);
    int kb = (l & 16) ? 16 : 0;
#pragma unroll
    for (int h = 0; h < 16; ++h) {
      int K = kb + h;
      wf[((size_t)(t * 2 + mt) * 32 + l) * 16 + h] =
          f2bf(w[((size_t)m * CC + K) * 27 + t]);    // [O][C][3][3][3]
    }
  } else if (idx < 3584) {  // point_w: 2 kchunks * 2 mtiles * 32 lanes
    int i = idx - 3456;
    int l = i & 31, mt = (i >> 5) & 1, kc = i >> 6;
    int m = mt * 16 + (l & 15);
    int kb = (l & 16) ? 16 : 0;
#pragma unroll
    for (int h = 0; h < 16; ++h) {
      int K = kb + h;
      pwf[((size_t)(kc * 2 + mt) * 32 + l) * 16 + h] =
          f2bf(pw[(size_t)m * (2 * CC) + kc * 32 + K]);  // [O][2C]
    }
  }
}

// ---------------- implicit-GEMM 3x3x3 conv via bf16 WMMA ----------------
// grid channels-last [b][vx][vy][vz][c]; one wave = 16 voxels (z-run) x 32 out ch.
__global__ void __launch_bounds__(128) k_conv(const unsigned short* __restrict__ in,
                                              const unsigned short* __restrict__ wf,
                                              const float* __restrict__ bias,
                                              const float* __restrict__ g,
                                              const float* __restrict__ bb,
                                              unsigned short* __restrict__ out_bf,
                                              float* __restrict__ out_f) {
  int wid = blockIdx.x * 4 + (threadIdx.x >> 5);
  int lane = threadIdx.x & 31;
  int b = wid >> 11;          // 2048 tiles per batch
  int t = wid & 2047;
  int vx = t >> 6, vy = (t >> 1) & 31, z0 = (t & 1) * 16;
  int nl = lane & 15;
  bool hi = lane >= 16;
  int koff = hi ? 16 : 0;

  v8f acc[2];
  acc[0] = {}; acc[1] = {};
  v16bf bz = {};

  for (int tt = 0; tt < 27; ++tt) {
    int kd = tt / 9, kh = (tt / 3) % 3, kw = tt % 3;
    int nx = vx + kd - 1, ny = vy + kh - 1;
    if ((unsigned)nx >= (unsigned)RR || (unsigned)ny >= (unsigned)RR) continue;  // zero pad
    int vz = z0 + nl + (kw - 1);
    bool ok = (unsigned)vz < (unsigned)RR;
    const unsigned short* bp =
        in + (((size_t)b * R3 + ((size_t)nx * RR + ny) * RR + (ok ? vz : 0)) * CC + koff);
    v16bf bfrag = ok ? *(const v16bf*)bp : bz;
    const unsigned short* ap = wf + ((size_t)(tt * 2) * 32 + lane) * 16;
    v16bf a0 = *(const v16bf*)ap;
    v16bf a1 = *(const v16bf*)(ap + 32 * 16);
    acc[0] = wmma_bf16(a0, bfrag, acc[0]);
    acc[1] = wmma_bf16(a1, bfrag, acc[1]);
  }

  size_t vvox = (size_t)b * R3 + ((size_t)vx * RR + vy) * RR + z0 + nl;
  float bnr = rsqrtf(1.0f + EPSBN);
#pragma unroll
  for (int mt = 0; mt < 2; ++mt) {
    int obase = mt * 16 + (hi ? 8 : 0);
    float r[8];
#pragma unroll
    for (int j = 0; j < 8; ++j) {
      int o = obase + j;
      float v = acc[mt][j] + bias[o];
      v = v * (g[o] * bnr) + bb[o];
      r[j] = (v > 0.0f) ? v : v * NEGS;           // LeakyReLU
    }
    if (out_bf) {
      us8 o8;
#pragma unroll
      for (int j = 0; j < 8; ++j) o8[j] = f2bf(r[j]);
      *(us8*)(out_bf + vvox * CC + obase) = o8;
    }
    if (out_f) {
      v4f lo, hs;
#pragma unroll
      for (int j = 0; j < 4; ++j) { lo[j] = r[j]; hs[j] = r[j + 4]; }
      *(v4f*)(out_f + vvox * CC + obase) = lo;
      *(v4f*)(out_f + vvox * CC + obase + 4) = hs;
    }
  }
}

// ---------------- fused devoxelize + point GEMM + add ----------------
__global__ void __launch_bounds__(256) k_point(const float* __restrict__ feat,
                                               const float* __restrict__ normc,
                                               const int* __restrict__ inds,
                                               const float* __restrict__ avg,
                                               const float* __restrict__ h2,
                                               const unsigned short* __restrict__ pwf,
                                               const float* __restrict__ pb,
                                               const float* __restrict__ pg,
                                               const float* __restrict__ pbb,
                                               float* __restrict__ out) {
  __shared__ __align__(32) unsigned short pin[8][32][64];  // point_in, bf16
  __shared__ float vfs[8][32][32];                         // devox result, f32
  int w = threadIdx.x >> 5, lane = threadIdx.x & 31;
  int gw = blockIdx.x * 8 + w;
  int b = gw >> 11;                // 2048 waves per batch (32 pts each)
  int p0 = (gw & 2047) * 32;
  int p = p0 + lane;               // lane <-> point for the gather phase

  float nx = normc[((size_t)b * 3 + 0) * NN + p];
  float ny = normc[((size_t)b * 3 + 1) * NN + p];
  float nz = normc[((size_t)b * 3 + 2) * NN + p];
  int ind = inds[(size_t)b * NN + p];

  // related = features - center ; stage concat(related, features) as bf16 in LDS
  const float* cb = avg + ((size_t)b * R3 + ind) * CC;
#pragma unroll 4
  for (int c = 0; c < CC; ++c) {
    float fv = feat[((size_t)b * CC + c) * NN + p];
    pin[w][lane][c]      = f2bf(fv - cb[c]);
    pin[w][lane][CC + c] = f2bf(fv);
  }

  // trilinear devoxelize from conv2 output (fp32, channels-last)
  float x0 = floorf(nx), y0 = floorf(ny), z0 = floorf(nz);
  int ix0 = (int)x0, iy0 = (int)y0, iz0 = (int)z0;
  int ix1 = min(ix0 + 1, RR - 1), iy1 = min(iy0 + 1, RR - 1), iz1 = min(iz0 + 1, RR - 1);
  float fx = nx - x0, fy = ny - y0, fz = nz - z0;
  float accv[CC];
#pragma unroll
  for (int c = 0; c < CC; ++c) accv[c] = 0.0f;
#pragma unroll
  for (int corner = 0; corner < 8; ++corner) {
    int cx = (corner & 4) ? ix1 : ix0;
    int cy = (corner & 2) ? iy1 : iy0;
    int cz = (corner & 1) ? iz1 : iz0;
    float wgt = ((corner & 4) ? fx : 1.0f - fx) * ((corner & 2) ? fy : 1.0f - fy) *
                ((corner & 1) ? fz : 1.0f - fz);
    const v4f* hp = (const v4f*)(h2 + ((size_t)b * R3 + ((size_t)cx * RR + cy) * RR + cz) * CC);
#pragma unroll
    for (int q = 0; q < 8; ++q) {
      v4f x = hp[q];
#pragma unroll
      for (int j = 0; j < 4; ++j) accv[q * 4 + j] += wgt * x[j];
    }
  }
#pragma unroll 4
  for (int c = 0; c < CC; ++c) vfs[w][lane][c] = accv[c];
  __syncthreads();

  // point GEMM: D[32 x 16pts] = point_w[32 x 64] @ point_in[64 x 16pts]
  v16bf aw[2][2];
#pragma unroll
  for (int kc = 0; kc < 2; ++kc)
#pragma unroll
    for (int mt = 0; mt < 2; ++mt)
      aw[kc][mt] = *(const v16bf*)(pwf + ((size_t)(kc * 2 + mt) * 32 + lane) * 16);

  int nl = lane & 15;
  bool hi = lane >= 16;
  float bnr = rsqrtf(1.0f + EPSBN);
#pragma unroll
  for (int s = 0; s < 2; ++s) {           // two 16-point subtiles
    int pp = s * 16 + nl;
    v8f d[2];
    d[0] = {}; d[1] = {};
#pragma unroll
    for (int kc = 0; kc < 2; ++kc) {
      v16bf bfrag = *(const v16bf*)&pin[w][pp][kc * 32 + (hi ? 16 : 0)];
      d[0] = wmma_bf16(aw[kc][0], bfrag, d[0]);
      d[1] = wmma_bf16(aw[kc][1], bfrag, d[1]);
    }
#pragma unroll
    for (int mt = 0; mt < 2; ++mt) {
      int obase = mt * 16 + (hi ? 8 : 0);
#pragma unroll
      for (int j = 0; j < 8; ++j) {
        int o = obase + j;
        float v = d[mt][j] + pb[o];
        v = v * (pg[o] * bnr) + pbb[o];
        v = fmaxf(v, 0.0f);               // ReLU
        v += vfs[w][pp][o];               // + devoxelized voxel features
        out[((size_t)b * OO + o) * NN + p0 + s * 16 + nl] = v;
      }
    }
  }
}

extern "C" void kernel_launch(void* const* d_in, const int* in_sizes, int n_in,
                              void* d_out, int out_size, void* d_ws, size_t ws_size,
                              hipStream_t stream) {
  (void)in_sizes; (void)n_in; (void)out_size; (void)ws_size;
  const float* features = (const float*)d_in[0];
  const float* coords   = (const float*)d_in[1];
  const float* w1  = (const float*)d_in[2];
  const float* b1  = (const float*)d_in[3];
  const float* g1  = (const float*)d_in[4];
  const float* bb1 = (const float*)d_in[5];
  const float* w2  = (const float*)d_in[6];
  const float* b2  = (const float*)d_in[7];
  const float* g2  = (const float*)d_in[8];
  const float* bb2 = (const float*)d_in[9];
  const float* pw  = (const float*)d_in[10];
  const float* pb  = (const float*)d_in[11];
  const float* pg  = (const float*)d_in[12];
  const float* pbb = (const float*)d_in[13];

  char* ws = (char*)d_ws;
  float*          normc = (float*)(ws + OFF_NORMC);
  float*          mean  = (float*)(ws + OFF_MEAN);
  float*          scale = (float*)(ws + OFF_SCALE);
  float*          psum  = (float*)(ws + OFF_PSUM);
  float*          pmax  = (float*)(ws + OFF_PMAX);
  int*            inds  = (int*)(ws + OFF_INDS);
  float*          sums  = (float*)(ws + OFF_SUMS);   // becomes avg (fp32) in place
  float*          cnt   = (float*)(ws + OFF_CNT);
  unsigned short* avgbf = (unsigned short*)(ws + OFF_AVGBF);
  unsigned short* h1    = (unsigned short*)(ws + OFF_H1);
  float*          h2    = (float*)(ws + OFF_H2);
  unsigned short* wf1   = (unsigned short*)(ws + OFF_WF1);
  unsigned short* wf2   = (unsigned short*)(ws + OFF_WF2);
  unsigned short* pwf   = (unsigned short*)(ws + OFF_PWF);
  float* out = (float*)d_out;

  // 1) coord stats
  k_mean_partial<<<BB * NBLK, 256, 0, stream>>>(coords, psum);
  k_mean_final<<<BB, 256, 0, stream>>>(psum, mean);
  k_rad_partial<<<BB * NBLK, 256, 0, stream>>>(coords, mean, pmax);
  k_rad_final<<<BB, 256, 0, stream>>>(pmax, scale);
  // 2) zero scatter targets (sums + cnt are contiguous)
  hipMemsetAsync(sums, 0, (size_t)BB * R3 * CC * 4 + (size_t)BB * R3 * 4, stream);
  // 3) voxelize (scatter-add)
  k_voxelize<<<(BB * NN) / 256, 256, 0, stream>>>(coords, features, mean, scale,
                                                  normc, inds, sums, cnt);
  // 4) scatter-mean finalize -> avg (fp32) + bf16 copy
  k_finalize_avg<<<((size_t)BB * R3 * CC) / 256, 256, 0, stream>>>(sums, cnt, avgbf);
  // 5) pack all weights into WMMA A-fragment layout
  k_pack<<<14, 256, 0, stream>>>(w1, w2, pw, wf1, wf2, pwf);
  // 6) conv1: bf16 WMMA, LeakyReLU, bf16 out
  k_conv<<<(BB * 2048) / 4, 128, 0, stream>>>(avgbf, wf1, b1, g1, bb1, h1, (float*)nullptr);
  // 7) conv2: bf16 WMMA, LeakyReLU, fp32 out (for trilinear devox)
  k_conv<<<(BB * 2048) / 4, 128, 0, stream>>>(h1, wf2, b2, g2, bb2, (unsigned short*)nullptr, h2);
  // 8) fused devoxelize + point GEMM (WMMA) + sum
  k_point<<<(BB * NN / 32) / 8, 256, 0, stream>>>(features, normc, inds, sums, h2,
                                                  pwf, pb, pg, pbb, out);
}
